// MultiHeadAttentionRoPE_89051851915924
// MI455X (gfx1250) — matrix operand order repaired and checked by reading, not compiled, not run on hardware
//
#include <hip/hip_runtime.h>
#include <hip/hip_bf16.h>

#define DMODEL 1024
#define SEQ    2048
#define NB     2
#define NH     16
#define HD     64
#define MTOT   (NB * SEQ)   // 4096
#define KSTEPS (DMODEL / 32)

typedef __attribute__((ext_vector_type(16))) __bf16 v16bf;
typedef __attribute__((ext_vector_type(8)))  float  v8f;
typedef __attribute__((ext_vector_type(4))) unsigned int u32x4;
typedef __attribute__((ext_vector_type(8))) int i32x8;
typedef __attribute__((ext_vector_type(4))) int i32x4;

union Frag16 { v16bf v; unsigned int u[8]; unsigned short s[16]; };

#if defined(__has_builtin)
#if __has_builtin(__builtin_amdgcn_tensor_load_to_lds)
#define USE_TDM 1
#endif
#endif

__device__ __forceinline__ unsigned short f32_to_bf16(float f) {
  unsigned int u = __float_as_uint(f);
  u += 0x7FFFu + ((u >> 16) & 1u);          // round-to-nearest-even
  return (unsigned short)(u >> 16);
}

// --- DPP16 ROW_XMASK butterfly: xor-permute within each row of 16 lanes ----
template <int XMASK>
__device__ __forceinline__ float dpp_xor16(float v) {
  return __int_as_float(__builtin_amdgcn_update_dpp(
      0, __float_as_int(v), 0x160 | XMASK, 0xF, 0xF, true));
}
__device__ __forceinline__ float rowmax16(float v) {
  v = fmaxf(v, dpp_xor16<1>(v));
  v = fmaxf(v, dpp_xor16<2>(v));
  v = fmaxf(v, dpp_xor16<4>(v));
  v = fmaxf(v, dpp_xor16<8>(v));
  return v;
}
__device__ __forceinline__ float rowsum16(float v) {
  v += dpp_xor16<1>(v);
  v += dpp_xor16<2>(v);
  v += dpp_xor16<4>(v);
  v += dpp_xor16<8>(v);
  return v;
}

#ifdef USE_TDM
// Issue one TDM 2D tile load (bf16 elements): D# groups per ISA 08 §8.3/8.4.
__device__ __forceinline__ void tdm_load_2d(const void* gaddr, unsigned lds_off,
                                            unsigned tensor_w, unsigned tensor_h,
                                            unsigned tile_w, unsigned tile_h) {
  unsigned long long ga = (unsigned long long)gaddr;
  u32x4 g0;
  g0[0] = 1u;                                             // count=1, flags=0
  g0[1] = lds_off;                                        // lds_addr (bytes)
  g0[2] = (unsigned)(ga & 0xFFFFFFFFu);                   // global_addr[31:0]
  g0[3] = (unsigned)((ga >> 32) & 0x01FFFFFFu) | (2u << 30); // addr[56:32]|type=2
  i32x8 g1;
  g1[0] = (int)(1u << 16);                                // data_size=1 -> 2 bytes
  g1[1] = (int)((tensor_w & 0xFFFFu) << 16);              // tensor_dim0[15:0]
  g1[2] = (int)(((tensor_w >> 16) & 0xFFFFu) | ((tensor_h & 0xFFFFu) << 16));
  g1[3] = (int)(((tensor_h >> 16) & 0xFFFFu) | ((tile_w & 0xFFFFu) << 16));
  g1[4] = (int)(tile_h & 0xFFFFu);                        // tile_dim1; tile_dim2=0
  g1[5] = (int)tensor_w;                                  // tensor_dim0_stride lo
  g1[6] = 0;                                              // stride hi, dim1_stride lo
  g1[7] = 0;
  i32x4 z4 = {0, 0, 0, 0};
#if defined(__clang_major__) && __clang_major__ >= 23
  i32x8 z8 = {0, 0, 0, 0, 0, 0, 0, 0};
  __builtin_amdgcn_tensor_load_to_lds(g0, g1, z4, z4, z8, 0);
#else
  __builtin_amdgcn_tensor_load_to_lds(g0, g1, z4, z4, 0);
#endif
}
#endif

__global__ void cast_f32_to_bf16_kernel(const float* __restrict__ in,
                                        unsigned short* __restrict__ out, int n) {
  int i = blockIdx.x * blockDim.x + threadIdx.x;
  if (i < n) out[i] = f32_to_bf16(in[i]);
}

// Transposing cast: in [K=1024][N=1024] f32 -> out [N][K] bf16 (W^T).
__global__ void cast_transpose_kernel(const float* __restrict__ in,
                                      unsigned short* __restrict__ out) {
  int i = blockIdx.x * blockDim.x + threadIdx.x;
  int k = i >> 10, n = i & (DMODEL - 1);
  out[(size_t)n * DMODEL + k] = f32_to_bf16(in[i]);
}

// -------------------------------------------------------------------------
// Tiled bf16 GEMM: C[4096,1024] = A[4096,1024] x W[1024,1024] + bias,
// weights given TRANSPOSED (WT = W^T, [N][K]) so both LDS tiles are plain
// row-major 2D tile copies (TDM-friendly) and all fragment K-pairs are
// contiguous dwords. 8 waves; block tile 128(M) x 64(N); wave: 16 x 64.
// Double-buffered tiles staged by the Tensor Data Mover (wave 0).
// MODE 0: Q-proj (bias + RoPE + 1/sqrt(d) scale, bf16 (B,H,S,D))
// MODE 1: K-proj (bias + RoPE,                    bf16 (B,H,S,D))
// MODE 2: V-proj (bias,                           bf16 (B,H,D,S) = V^T)
// MODE 3: O-proj (bias,                           f32 row-major -> d_out)
// -------------------------------------------------------------------------
template <int MODE>
__global__ __launch_bounds__(256)
void gemm16x64_kernel(const unsigned short* __restrict__ A,
                      const unsigned short* __restrict__ WT,
                      const float* __restrict__ bias,
                      unsigned short* __restrict__ outB,
                      float* __restrict__ outF) {
  __shared__ unsigned int lA[2][128 * 16];  // 128 rows x 32 k (dword = K-pair)
  __shared__ unsigned int lB[2][64 * 16];   // 64 n-rows x 32 k

  const int tid  = threadIdx.x;
  const int wid  = tid >> 5;
  const int lane = tid & 31;
  const int hl   = lane >> 4;   // K-half / M-half selector
  const int ml   = lane & 15;

  const int m0   = blockIdx.y * 128;
  const int n0   = blockIdx.x * 64;
  const int head = blockIdx.x;              // 64-wide N tile == one head

  v8f acc[4];
#pragma unroll
  for (int f = 0; f < 4; ++f)
#pragma unroll
    for (int r = 0; r < 8; ++r) acc[f][r] = 0.0f;

#ifdef USE_TDM
  const unsigned ldsA0 = (unsigned)(unsigned long long)(uintptr_t)&lA[0][0];
  const unsigned ldsB0 = (unsigned)(unsigned long long)(uintptr_t)&lB[0][0];
#endif

  // ---- stage tiles for step 'st' into buffer 'buf' ----
  auto stage = [&](int st, int buf) {
    const int k0 = st * 32;
#ifdef USE_TDM
    if (tid < 32) {   // wave 0 issues both DMAs; TDM ignores EXEC
      tdm_load_2d(A + (size_t)m0 * DMODEL + k0, ldsA0 + buf * (128 * 64),
                  DMODEL, MTOT, 32, 128);
      tdm_load_2d(WT + (size_t)n0 * DMODEL + k0, ldsB0 + buf * (64 * 64),
                  DMODEL, DMODEL, 32, 64);
    }
#else
#pragma unroll
    for (int t = tid; t < 2048; t += 256) {
      int row = t >> 4, c = t & 15;
      lA[buf][t] = *(const unsigned int*)(A + (size_t)(m0 + row) * DMODEL + k0 + 2 * c);
    }
#pragma unroll
    for (int t = tid; t < 1024; t += 256) {
      int row = t >> 4, c = t & 15;
      lB[buf][t] = *(const unsigned int*)(WT + (size_t)(n0 + row) * DMODEL + k0 + 2 * c);
    }
#endif
  };

  stage(0, 0);
  for (int st = 0; st < KSTEPS; ++st) {
    const int buf = st & 1;
    if (st + 1 < KSTEPS) stage(st + 1, buf ^ 1);
#ifdef USE_TDM
    if (tid < 32) {  // in-order TENSORcnt: <=2 => this buffer's pair landed
      if (st + 1 < KSTEPS) __builtin_amdgcn_s_wait_tensorcnt(2);
      else                 __builtin_amdgcn_s_wait_tensorcnt(0);
    }
#endif
    __syncthreads();

    // A fragment: lane holds row (wid*16+ml); VGPR i -> K pair per ISA 7.12.2
    Frag16 af;
#pragma unroll
    for (int i = 0; i < 8; ++i)
      af.u[i] = lA[buf][(wid * 16 + ml) * 16 + ((i >> 2) * 8 + hl * 4 + (i & 3))];

#pragma unroll
    for (int nf = 0; nf < 4; ++nf) {
      Frag16 bf;     // B VGPR v: K-pair (v>>2)*8+4*hl+(v&3) of output column nf*16+ml
#pragma unroll
      for (int v = 0; v < 8; ++v)
        bf.u[v] = lB[buf][(nf * 16 + ml) * 16 + ((v >> 2) * 8 + hl * 4 + (v & 3))];
      acc[nf] = __builtin_amdgcn_wmma_f32_16x16x32_bf16(
          false, af.v, false, bf.v, (short)0, acc[nf], false, false);
    }
    __syncthreads();
  }

  // ---- epilogue: bias (+RoPE/scale) + store ----
  if (MODE == 0 || MODE == 1) {
    const float qs = (MODE == 0) ? 0.125f : 1.0f;   // fold 1/sqrt(64) into Q
#pragma unroll
    for (int nf = 0; nf < 2; ++nf) {
      int d = nf * 16 + ml;                                  // d in [0,32)
      float invf = __expf(-(float)d * 0.28782313662425575f); // 10000^(-d/32)
      float b1 = bias[n0 + d];
      float b2 = bias[n0 + d + 32];
#pragma unroll
      for (int r = 0; r < 8; ++r) {
        int gm   = m0 + wid * 16 + r + 8 * hl;
        int srow = gm & (SEQ - 1);
        int bidx = gm >> 11;
        float sn, cs;
        __sincosf((float)srow * invf, &sn, &cs);
        float v1 = acc[nf][r] + b1;
        float v2 = acc[nf + 2][r] + b2;
        size_t base = (((size_t)(bidx * NH + head)) * SEQ + srow) * HD;
        outB[base + d]      = f32_to_bf16((v1 * cs - v2 * sn) * qs);
        outB[base + d + 32] = f32_to_bf16((v2 * cs + v1 * sn) * qs);
      }
    }
  } else if (MODE == 2) {
#pragma unroll
    for (int nf = 0; nf < 4; ++nf) {
      int d = nf * 16 + ml;
      float bb = bias[n0 + d];
#pragma unroll
      for (int r = 0; r < 8; ++r) {
        int gm   = m0 + wid * 16 + r + 8 * hl;
        int srow = gm & (SEQ - 1);
        int bidx = gm >> 11;
        size_t base = (((size_t)(bidx * NH + head)) * HD + d) * SEQ;
        outB[base + srow] = f32_to_bf16(acc[nf][r] + bb);    // V^T (B,H,D,S)
      }
    }
  } else {
#pragma unroll
    for (int nf = 0; nf < 4; ++nf) {
      float bb = bias[n0 + nf * 16 + ml];
#pragma unroll
      for (int r = 0; r < 8; ++r) {
        int gm = m0 + wid * 16 + r + 8 * hl;
        outF[(size_t)gm * DMODEL + n0 + nf * 16 + ml] = acc[nf][r] + bb;
      }
    }
  }
}

// -------------------------------------------------------------------------
// Flash attention: one wave per 16-query tile, 32 keys per step.
// Q,K in (B,H,S,D) bf16 (Q pre-scaled by 1/sqrt(d)); V in (B,H,D,S) bf16.
// -------------------------------------------------------------------------
__global__ __launch_bounds__(256)
void flash_attn_kernel(const unsigned short* __restrict__ Q,
                       const unsigned short* __restrict__ K,
                       const unsigned short* __restrict__ VT,
                       unsigned short* __restrict__ O) {
  __shared__ unsigned short lp[8 * 16 * 32];   // per-wave P staging (1 KB each)

  const int tid  = threadIdx.x;
  const int wid  = tid >> 5;
  const int lane = tid & 31;
  const int hl   = lane >> 4;
  const int ml   = lane & 15;
  const int bh   = blockIdx.y;
  const int b    = bh >> 4, head = bh & 15;
  const int q0   = (blockIdx.x * 8 + wid) * 16;

  // Q fragments (16 x 64 -> two K=32 A-fragments), resident for whole loop
  Frag16 qf[2];
  const unsigned int* qrow =
      (const unsigned int*)(Q + ((size_t)bh * SEQ + q0 + ml) * HD);
#pragma unroll
  for (int kc = 0; kc < 2; ++kc)
#pragma unroll
    for (int i = 0; i < 8; ++i)
      qf[kc].u[i] = qrow[kc * 16 + ((i >> 2) * 8 + hl * 4 + (i & 3))];

  float mrow[8], lrow[8];
  v8f o[4];
#pragma unroll
  for (int r = 0; r < 8; ++r) { mrow[r] = -3.0e38f; lrow[r] = 0.0f; }
#pragma unroll
  for (int f = 0; f < 4; ++f)
#pragma unroll
    for (int r = 0; r < 8; ++r) o[f][r] = 0.0f;

  unsigned short* myp = &lp[wid * 512];
  const unsigned int* myp32 = (const unsigned int*)myp;

  for (int kv0 = 0; kv0 < SEQ; kv0 += 32) {
    // scores S = Q K^T (two 16x16 fragments; depth 64 = 2 chained WMMAs each)
    v8f sc[2];
#pragma unroll
    for (int nf = 0; nf < 2; ++nf) {
      Frag16 kb0, kb1;
      const unsigned int* krow =
          (const unsigned int*)(K + ((size_t)bh * SEQ + kv0 + nf * 16 + ml) * HD);
#pragma unroll
      for (int v = 0; v < 8; ++v) {
        kb0.u[v] = krow[(v >> 2) * 8 + hl * 4 + (v & 3)];        // depth 0..31
        kb1.u[v] = krow[16 + (v >> 2) * 8 + hl * 4 + (v & 3)];   // depth 32..63
      }
      v8f z;
#pragma unroll
      for (int r = 0; r < 8; ++r) z[r] = 0.0f;
      sc[nf] = __builtin_amdgcn_wmma_f32_16x16x32_bf16(false, qf[0].v, false, kb0.v, (short)0, z, false, false);
      sc[nf] = __builtin_amdgcn_wmma_f32_16x16x32_bf16(false, qf[1].v, false, kb1.v, (short)0, sc[nf], false, false);
    }

    // online softmax: rows live in one 16-lane half -> DPP16 ROW_XMASK
#pragma unroll
    for (int r = 0; r < 8; ++r) {
      float s0 = sc[0][r];
      float s1 = sc[1][r];
      float mx = rowmax16(fmaxf(s0, s1));
      float mn = fmaxf(mrow[r], mx);
      float al = __expf(mrow[r] - mn);
      float p0 = __expf(s0 - mn);
      float p1 = __expf(s1 - mn);
      float rs = rowsum16(p0 + p1);
      lrow[r] = lrow[r] * al + rs;
      mrow[r] = mn;
      sc[0][r] = p0;
      sc[1][r] = p1;
      o[0][r] *= al; o[1][r] *= al; o[2][r] *= al; o[3][r] *= al;
    }

    // restage P (C layout -> row-major LDS -> A-fragment layout); wave-private,
    // DScnt ordering covers the store->load dependency.
#pragma unroll
    for (int nf = 0; nf < 2; ++nf)
#pragma unroll
      for (int r = 0; r < 8; ++r)
        myp[(r + 8 * hl) * 32 + nf * 16 + ml] = f32_to_bf16(sc[nf][r]);

    Frag16 pf;
#pragma unroll
    for (int i = 0; i < 8; ++i)
      pf.u[i] = myp32[ml * 16 + ((i >> 2) * 8 + hl * 4 + (i & 3))];

    // O += P V : V^T layout makes each B-fragment VGPR one contiguous dword
#pragma unroll
    for (int f4 = 0; f4 < 4; ++f4) {
      Frag16 vb;
      const unsigned int* vrow =
          (const unsigned int*)(VT + (((size_t)bh * HD) + f4 * 16 + ml) * SEQ + kv0);
#pragma unroll
      for (int v = 0; v < 8; ++v) vb.u[v] = vrow[(v >> 2) * 8 + hl * 4 + (v & 3)];
      o[f4] = __builtin_amdgcn_wmma_f32_16x16x32_bf16(false, pf.v, false, vb.v, (short)0, o[f4], false, false);
    }
  }

  // normalize + store (B, S, DMODEL) bf16 for the output projection
#pragma unroll
  for (int f4 = 0; f4 < 4; ++f4)
#pragma unroll
    for (int r = 0; r < 8; ++r) {
      int srow = q0 + r + 8 * hl;
      int d    = f4 * 16 + ml;
      float val = o[f4][r] / lrow[r];
      O[((size_t)(b * SEQ + srow)) * DMODEL + head * HD + d] = f32_to_bf16(val);
    }
}

// -------------------------------------------------------------------------
extern "C" void kernel_launch(void* const* d_in, const int* in_sizes, int n_in,
                              void* d_out, int out_size, void* d_ws, size_t ws_size,
                              hipStream_t stream) {
  (void)in_sizes; (void)n_in; (void)out_size; (void)ws_size;
  const float* x  = (const float*)d_in[0];
  const float* wq = (const float*)d_in[1];
  const float* bq = (const float*)d_in[2];
  const float* wk = (const float*)d_in[3];
  const float* bk = (const float*)d_in[4];
  const float* wv = (const float*)d_in[5];
  const float* bv = (const float*)d_in[6];
  const float* wo = (const float*)d_in[7];
  const float* bo = (const float*)d_in[8];
  float* out = (float*)d_out;

  const size_t NX = (size_t)MTOT * DMODEL;      // 4,194,304
  const size_t NW = (size_t)DMODEL * DMODEL;    // 1,048,576
  unsigned short* xb  = (unsigned short*)d_ws;  // x bf16; later reused as attn out
  unsigned short* wqb = xb  + NX;               // all weights stored transposed
  unsigned short* wkb = wqb + NW;
  unsigned short* wvb = wkb + NW;
  unsigned short* wob = wvb + NW;
  unsigned short* Qb  = wob + NW;               // (B,H,S,D)
  unsigned short* Kb  = Qb  + NX;               // (B,H,S,D)
  unsigned short* Vb  = Kb  + NX;               // (B,H,D,S)  total ~40 MB

  cast_f32_to_bf16_kernel<<<(int)((NX + 255) / 256), 256, 0, stream>>>(x, xb, (int)NX);
  cast_transpose_kernel<<<(int)(NW / 256), 256, 0, stream>>>(wq, wqb);
  cast_transpose_kernel<<<(int)(NW / 256), 256, 0, stream>>>(wk, wkb);
  cast_transpose_kernel<<<(int)(NW / 256), 256, 0, stream>>>(wv, wvb);
  cast_transpose_kernel<<<(int)(NW / 256), 256, 0, stream>>>(wo, wob);

  dim3 gg(DMODEL / 64, MTOT / 128);             // (16, 32)
  gemm16x64_kernel<0><<<gg, 256, 0, stream>>>(xb, wqb, bq, Qb, nullptr);
  gemm16x64_kernel<1><<<gg, 256, 0, stream>>>(xb, wkb, bk, Kb, nullptr);
  gemm16x64_kernel<2><<<gg, 256, 0, stream>>>(xb, wvb, bv, Vb, nullptr);

  dim3 ga(SEQ / 128, NB * NH);                  // (16, 32): 8 q-tiles per block
  flash_attn_kernel<<<ga, 256, 0, stream>>>(Qb, Kb, Vb, xb);   // xb <- attn out

  gemm16x64_kernel<3><<<gg, 256, 0, stream>>>(xb, wob, bo, nullptr, out);
}
